// SparseAttention_90529320665341
// MI455X (gfx1250) — matrix operand order repaired and checked by reading, not compiled, not run on hardware
//
#include <hip/hip_runtime.h>

// ---------------------------------------------------------------------------
// Sparse (local-window) attention, MI455X / gfx1250.
// 4 big GEMMs via v_wmma_f32_16x16x32_bf16 (f32 accum), TDM tensor_load_to_lds
// double-buffered staging; banded softmax in VALU (0.4% of FLOPs).
// ---------------------------------------------------------------------------

#define S_LEN   2048
#define HEADS   16
#define DKK     64
#define DMODEL  1024
#define BATCH   2
#define MROWS   (BATCH * S_LEN)   // 4096
#define WINDOW  4

typedef __attribute__((ext_vector_type(16))) __bf16        v16bf;
typedef __attribute__((ext_vector_type(8)))  __bf16        v8bf;
typedef __attribute__((ext_vector_type(8)))  float         v8f;
typedef __attribute__((ext_vector_type(4)))  unsigned int  v4u;
typedef __attribute__((ext_vector_type(8)))  int           v8i;
typedef __attribute__((ext_vector_type(4)))  int           v4i;

union Frag { v16bf v; v8bf h[2]; };

static __device__ __forceinline__ unsigned short f2bf(float f) {
  unsigned int u = __float_as_uint(f);
  unsigned int r = u + 0x7fffu + ((u >> 16) & 1u);   // round-to-nearest-even
  return (unsigned short)(r >> 16);
}
static __device__ __forceinline__ float bf2f(unsigned int u) {
  return __uint_as_float(u << 16);
}

// ---------------- fp32 -> bf16 convert (vectorized) ------------------------
__global__ __launch_bounds__(256) void cvt_f32_to_bf16(
    const float* __restrict__ src, unsigned short* __restrict__ dst, int n4) {
  int i = blockIdx.x * 256 + threadIdx.x;
  if (i < n4) {
    float4 f = ((const float4*)src)[i];
    ushort4 o;
    o.x = f2bf(f.x); o.y = f2bf(f.y); o.z = f2bf(f.z); o.w = f2bf(f.w);
    ((ushort4*)dst)[i] = o;
  }
}

// ---------------- bf16 WMMA GEMM:  C[M,N] = A[M,K] * W[N,K]^T (+bias) ------
#define TM  128
#define TN  128
#define TK  32
#define LDP 40   // LDS row stride in halfwords (32 + 8 pad; TDM pad reproduces this)

#if defined(__HIP_DEVICE_COMPILE__) && __has_builtin(__builtin_amdgcn_tensor_load_to_lds)
#define USE_TDM 1
#else
#define USE_TDM 0
#endif

#if USE_TDM
// One TDM descriptor: load a [128 x 32] bf16 tile (row stride K elems) into LDS
// at byte offset lds_byte, padding 4 DWORDs after every 16 DWORDs (-> LDP=40).
static __device__ __forceinline__ void tdm_load_tile(
    const unsigned short* gsrc, unsigned lds_byte, int K) {
  unsigned long long ga = (unsigned long long)(size_t)gsrc;   // byte address
  v4u g0;
  g0.x = 1u;                                          // count=1 (valid user D#)
  g0.y = lds_byte;                                    // lds_addr
  g0.z = (unsigned)(ga & 0xffffffffu);                // global_addr[31:0]
  g0.w = (unsigned)((ga >> 32) & 0x01ffffffu) | (2u << 30);  // [56:32], type=2
  v8i g1;
  g1[0] = (1 << 16)        // data_size = 2 bytes
        | (1 << 20)        // pad_enable
        | (3 << 22)        // pad_interval: 16 DWORDs (one 64B row)
        | (3 << 25);       // pad_amount:   4 DWORDs (8 halfwords)
  g1[1] = (TK & 0xffff) << 16;        // tensor_dim0[15:0] = 32 (tile-relative)
  g1[2] = (TM & 0xffff) << 16;        // tensor_dim1[15:0] = 128
  g1[3] = (TK & 0xffff) << 16;        // tile_dim0 = 32
  g1[4] = TM;                         // tile_dim1 = 128, tile_dim2 = 0
  g1[5] = K;                          // tensor_dim0_stride[31:0]
  g1[6] = 0;                          // stride0 hi / dim1_stride lo (2D: unused)
  g1[7] = 0;
  v4i z4 = {0, 0, 0, 0};
#if __clang_major__ >= 23
  v8i z8 = {0, 0, 0, 0, 0, 0, 0, 0};
  __builtin_amdgcn_tensor_load_to_lds(g0, g1, z4, z4, z8, 0);
#else
  __builtin_amdgcn_tensor_load_to_lds(g0, g1, z4, z4, 0);
#endif
}
#endif

template <bool OUTF32, bool BIAS>
static __device__ __forceinline__ void gemm_tile(
    const unsigned short* __restrict__ A,   // [M,K] bf16 row-major
    const unsigned short* __restrict__ W,   // [N,K] bf16 row-major (= B^T)
    const float* __restrict__ bias,         // [N] (BIAS only)
    void* __restrict__ Cout,                // [M,N] f32 or bf16
    int N, int K)
{
  __shared__ __align__(16) unsigned short lA[2][TM * LDP];
  __shared__ __align__(16) unsigned short lB[2][TN * LDP];

  const int tid  = threadIdx.x;           // 128 threads = 4 waves
  const int lane = tid & 31;
  const int wave = tid >> 5;              // 0..3
  const int wm   = wave >> 1;             // 0..1 : 64-row strip
  const int wn   = wave & 1;              // 0..1 : 64-col strip
  const int hf   = lane >> 4;
  const int lr   = lane & 15;

  const int m0 = blockIdx.y * TM;
  const int n0 = blockIdx.x * TN;
  const unsigned short* Abase = A + (size_t)m0 * K;
  const unsigned short* Wbase = W + (size_t)n0 * K;

  v8f acc[4][4];
  #pragma unroll
  for (int i = 0; i < 4; ++i)
    #pragma unroll
    for (int j = 0; j < 4; ++j) { v8f z = {}; acc[i][j] = z; }

  const int nk = K / TK;

#if USE_TDM
  const unsigned la0 = (unsigned)(size_t)&lA[0][0];   // flat[31:0] == LDS addr
  const unsigned la1 = (unsigned)(size_t)&lA[1][0];
  const unsigned lb0 = (unsigned)(size_t)&lB[0][0];
  const unsigned lb1 = (unsigned)(size_t)&lB[1][0];
  if (wave == 0) {                        // one TDM producer wave per block
    tdm_load_tile(Abase, la0, K);
    tdm_load_tile(Wbase, lb0, K);
  }
#endif

  for (int kt = 0; kt < nk; ++kt) {
#if USE_TDM
    if (wave == 0) {
      if (kt + 1 < nk) {                  // prefetch next tile into alt buffer
        const size_t ko = (size_t)(kt + 1) * TK;
        tdm_load_tile(Abase + ko, ((kt + 1) & 1) ? la1 : la0, K);
        tdm_load_tile(Wbase + ko, ((kt + 1) & 1) ? lb1 : lb0, K);
        __builtin_amdgcn_s_wait_tensorcnt(2);   // current tile landed
      } else {
        __builtin_amdgcn_s_wait_tensorcnt(0);
      }
    }
    __syncthreads();                      // tile kt visible to all waves
#else
    {   // fallback: cooperative register-staged copy into padded layout
      const size_t ko = (size_t)kt * TK;
      __syncthreads();
      for (int c = tid; c < 1024; c += 128) {
        const int mat = c >> 9;           // 0 = A, 1 = W
        const int cc  = c & 511;          // row = cc>>2, kpart = (cc&3)*8
        const unsigned short* src = mat ? Wbase : Abase;
        uint4 v = *(const uint4*)(src + (size_t)(cc >> 2) * K + ko + (cc & 3) * 8);
        unsigned short* dst = mat ? &lB[kt & 1][0] : &lA[kt & 1][0];
        *(uint4*)(dst + (cc >> 2) * LDP + (cc & 3) * 8) = v;
      }
      __syncthreads();
    }
#endif

    const unsigned short* bufA = &lA[kt & 1][0];
    const unsigned short* bufB = &lB[kt & 1][0];

    // A fragment: lane holds row m, K = {hf*8..+7, 16+hf*8..+7} (ISA A layout)
    Frag fa[4], fb[4];
    #pragma unroll
    for (int i = 0; i < 4; ++i) {
      const int row = wm * 64 + i * 16 + lr;
      fa[i].h[0] = *(const v8bf*)(bufA + row * LDP + hf * 8);
      fa[i].h[1] = *(const v8bf*)(bufA + row * LDP + 16 + hf * 8);
    }
    // B fragment: lane holds column n (= W row n), K = {hf*8..+7, 16+hf*8..+7}
    #pragma unroll
    for (int j = 0; j < 4; ++j) {
      const int row = wn * 64 + j * 16 + lr;
      fb[j].h[0] = *(const v8bf*)(bufB + row * LDP + hf * 8);
      fb[j].h[1] = *(const v8bf*)(bufB + row * LDP + 16 + hf * 8);
    }
    #pragma unroll
    for (int i = 0; i < 4; ++i)
      #pragma unroll
      for (int j = 0; j < 4; ++j)
        acc[i][j] = __builtin_amdgcn_wmma_f32_16x16x32_bf16(
            false, fa[i].v, false, fb[j].v, (short)0, acc[i][j], false, false);

    __syncthreads();                      // all reads done before buffer reuse
  }

  // Epilogue: C/D layout — VGPR r, lane: row = r + hf*8, col = lr
  #pragma unroll
  for (int i = 0; i < 4; ++i) {
    #pragma unroll
    for (int j = 0; j < 4; ++j) {
      const int col  = n0 + wn * 64 + j * 16 + lr;
      const int row0 = m0 + wm * 64 + i * 16 + hf * 8;
      const float bv = BIAS ? bias[col] : 0.0f;
      #pragma unroll
      for (int r = 0; r < 8; ++r) {
        const float val = acc[i][j][r] + bv;
        const size_t idx = (size_t)(row0 + r) * N + col;
        if (OUTF32) ((float*)Cout)[idx] = val;
        else        ((unsigned short*)Cout)[idx] = f2bf(val);
      }
    }
  }
}

__global__ __launch_bounds__(128) void qkv_proj_kernel(
    const unsigned short* XQ, const unsigned short* XK, const unsigned short* XV,
    const unsigned short* WQ, const unsigned short* WK, const unsigned short* WV,
    unsigned short* QP, unsigned short* KP, unsigned short* VP) {
  const unsigned short *A, *W; unsigned short* O;
  if (blockIdx.z == 0)      { A = XQ; W = WQ; O = QP; }
  else if (blockIdx.z == 1) { A = XK; W = WK; O = KP; }
  else                      { A = XV; W = WV; O = VP; }
  gemm_tile<false, false>(A, W, nullptr, O, DMODEL, DMODEL);
}

__global__ __launch_bounds__(128) void out_proj_kernel(
    const unsigned short* A, const unsigned short* W, const float* bias, float* O) {
  gemm_tile<true, true>(A, W, bias, O, DMODEL, DMODEL);
}

// ---------------- banded local attention (|i-j| <= 4), one wave per row ----
__global__ __launch_bounds__(256) void local_attn_kernel(
    const unsigned short* __restrict__ Q, const unsigned short* __restrict__ K,
    const unsigned short* __restrict__ V, unsigned short* __restrict__ O) {
  const int lane = threadIdx.x & 31;
  const int wave = threadIdx.x >> 5;
  const int r  = blockIdx.x * 8 + wave;   // (b*H + h)*S + q
  const int q  = r & (S_LEN - 1);
  const int bh = r >> 11;
  const int h  = bh & (HEADS - 1);
  const int b  = bh >> 4;

  const size_t headOff = (size_t)h * DKK + 2 * lane;        // 2 dims per lane
  const size_t qIdx = ((size_t)(b * S_LEN + q)) * DMODEL + headOff;

  const unsigned int qp = *(const unsigned int*)(Q + qIdx);
  const float q0 = bf2f(qp & 0xffffu), q1 = bf2f(qp >> 16);

  float sc[9];
  #pragma unroll
  for (int jj = 0; jj < 9; ++jj) {
    const int j = q - WINDOW + jj;
    float s = -1.0e30f;
    if ((unsigned)j < (unsigned)S_LEN) {                    // wave-uniform
      const size_t kIdx = ((size_t)(b * S_LEN + j)) * DMODEL + headOff;
      const unsigned int kp = *(const unsigned int*)(K + kIdx);
      float p = q0 * bf2f(kp & 0xffffu) + q1 * bf2f(kp >> 16);
      #pragma unroll
      for (int off = 16; off > 0; off >>= 1) p += __shfl_xor(p, off, 32);
      s = p * 0.125f;                                       // 1/sqrt(dk=64)
    }
    sc[jj] = s;
  }

  float mx = sc[0];
  #pragma unroll
  for (int jj = 1; jj < 9; ++jj) mx = fmaxf(mx, sc[jj]);
  float den = 0.0f, w[9];
  #pragma unroll
  for (int jj = 0; jj < 9; ++jj) { w[jj] = __expf(sc[jj] - mx); den += w[jj]; }
  const float inv = 1.0f / den;

  float a0 = 0.0f, a1 = 0.0f;
  #pragma unroll
  for (int jj = 0; jj < 9; ++jj) {
    const int j = q - WINDOW + jj;
    if ((unsigned)j < (unsigned)S_LEN) {
      const size_t vIdx = ((size_t)(b * S_LEN + j)) * DMODEL + headOff;
      const unsigned int vp = *(const unsigned int*)(V + vIdx);
      a0 = fmaf(w[jj], bf2f(vp & 0xffffu), a0);
      a1 = fmaf(w[jj], bf2f(vp >> 16), a1);
    }
  }
  a0 *= inv; a1 *= inv;
  *(unsigned int*)(O + qIdx) =
      (unsigned int)f2bf(a0) | ((unsigned int)f2bf(a1) << 16);
}

// ---------------------------------------------------------------------------
extern "C" void kernel_launch(void* const* d_in, const int* in_sizes, int n_in,
                              void* d_out, int out_size, void* d_ws, size_t ws_size,
                              hipStream_t stream) {
  (void)in_sizes; (void)n_in; (void)out_size; (void)ws_size;
  const float* query = (const float*)d_in[0];
  const float* key   = (const float*)d_in[1];
  const float* value = (const float*)d_in[2];
  const float* Wq    = (const float*)d_in[3];
  const float* Wk    = (const float*)d_in[4];
  const float* Wv    = (const float*)d_in[5];
  const float* Wo    = (const float*)d_in[6];
  const float* bo    = (const float*)d_in[7];

  constexpr size_t NX = (size_t)MROWS * DMODEL;    // 4,194,304
  constexpr size_t NW = (size_t)DMODEL * DMODEL;   // 1,048,576

  unsigned short* XQ  = (unsigned short*)d_ws;     // 64 MB total staging
  unsigned short* XK  = XQ  + NX;
  unsigned short* XV  = XK  + NX;
  unsigned short* WQb = XV  + NX;
  unsigned short* WKb = WQb + NW;
  unsigned short* WVb = WKb + NW;
  unsigned short* WOb = WVb + NW;
  unsigned short* QP  = WOb + NW;
  unsigned short* KP  = QP  + NX;
  unsigned short* VP  = KP  + NX;
  unsigned short* AO  = VP  + NX;

  const dim3 blk256(256);
  cvt_f32_to_bf16<<<dim3((unsigned)(NX / 4 / 256)), blk256, 0, stream>>>(query, XQ, (int)(NX / 4));
  cvt_f32_to_bf16<<<dim3((unsigned)(NX / 4 / 256)), blk256, 0, stream>>>(key,   XK, (int)(NX / 4));
  cvt_f32_to_bf16<<<dim3((unsigned)(NX / 4 / 256)), blk256, 0, stream>>>(value, XV, (int)(NX / 4));
  cvt_f32_to_bf16<<<dim3((unsigned)(NW / 4 / 256)), blk256, 0, stream>>>(Wq, WQb, (int)(NW / 4));
  cvt_f32_to_bf16<<<dim3((unsigned)(NW / 4 / 256)), blk256, 0, stream>>>(Wk, WKb, (int)(NW / 4));
  cvt_f32_to_bf16<<<dim3((unsigned)(NW / 4 / 256)), blk256, 0, stream>>>(Wv, WVb, (int)(NW / 4));
  cvt_f32_to_bf16<<<dim3((unsigned)(NW / 4 / 256)), blk256, 0, stream>>>(Wo, WOb, (int)(NW / 4));

  const dim3 blk128(128);
  dim3 gQKV(DMODEL / TN, MROWS / TM, 3);           // 8 x 32 x 3 = 768 WGs
  qkv_proj_kernel<<<gQKV, blk128, 0, stream>>>(XQ, XK, XV, WQb, WKb, WVb, QP, KP, VP);

  local_attn_kernel<<<dim3((BATCH * HEADS * S_LEN) / 8), blk256, 0, stream>>>(QP, KP, VP, AO);

  dim3 gO(DMODEL / TN, MROWS / TM, 1);
  out_proj_kernel<<<gO, blk128, 0, stream>>>(AO, WOb, bo, (float*)d_out);
}
